// LongformerEncoder_10651518894338
// MI455X (gfx1250) — compile-verified, matrix operand order checked by reading, not run on hardware
//
#include <hip/hip_runtime.h>
#include <cstdint>
#include <cstddef>

// ---------------- constants (match reference) ----------------
constexpr int   kB    = 2;
constexpr int   kS    = 4096;
constexpr int   kD    = 768;
constexpr int   kH    = 12;
constexpr int   kDH   = 64;          // D / H
constexpr int   kL    = 2;
constexpr int   kFF   = 3072;
constexpr int   kW    = 256;         // W2
constexpr int   kNSEP = 8;
constexpr int   kG    = kNSEP + 1;   // 9
constexpr int   kSEP  = 2;
constexpr float kScale = 0.125f;     // 1/sqrt(64)
constexpr float kEps   = 1e-5f;

// ---------------- WMMA types ----------------
typedef __attribute__((ext_vector_type(16))) __bf16 bf16x16;
typedef __attribute__((ext_vector_type(8)))  float  f32x8;

struct alignas(16) U128 { unsigned int x0, x1, x2, x3; };
union BFVec { bf16x16 v; U128 u[2]; };

// ---------------- helpers ----------------
__device__ inline unsigned short f2bf(float f) {
  unsigned int u = __float_as_uint(f);
  u += 0x7FFFu + ((u >> 16) & 1u);     // round-to-nearest-even
  return (unsigned short)(u >> 16);
}

__device__ inline float waveSum(float v) {
#pragma unroll
  for (int o = 16; o > 0; o >>= 1) v += __shfl_xor(v, o, 32);
  return v;
}

__device__ inline float blockSum(float v, float* sm) {
  v = waveSum(v);
  __syncthreads();
  if ((threadIdx.x & 31) == 0) sm[threadIdx.x >> 5] = v;
  __syncthreads();
  float t = 0.f;
#pragma unroll
  for (int i = 0; i < 8; ++i) t += sm[i];
  return t;
}

__device__ inline float gelu_tanh(float x) {
  float x3 = x * x * x;
  return 0.5f * x * (1.f + tanhf(0.7978845608028654f * (x + 0.044715f * x3)));
}

// ---------------- bf16 WMMA GEMM: C = act(scale*(A @ Bt^T + bias)) ----------------
// A  : [M,K] bf16 row-major
// Bt : [N,K] bf16 (weight pre-transposed so K is contiguous)
// Register-blocked: one wave -> 16x64 output (1 A fragment, 4 B fragments,
// 4 WMMA accumulators) -> 4 v_wmma per 10 b128 loads per k-step.
// Requires N % 64 == 0 (N is 768 or 3072 here).
__global__ __launch_bounds__(256) void gemm_bf16(
    const unsigned short* __restrict__ A,
    const unsigned short* __restrict__ Bt,
    const float* __restrict__ bias,
    float* __restrict__ Cf, unsigned short* __restrict__ Cb,
    int M, int N, int K, float scale, int act)
{
  const int tilesN4 = N >> 6;                 // 64-wide N tiles
  const int tilesM  = (M + 15) >> 4;
  const int wid = blockIdx.x * 8 + (threadIdx.x >> 5);
  if (wid >= tilesM * tilesN4) return;
  const int tm = wid / tilesN4;
  const int tn = wid - tm * tilesN4;
  const int lane = threadIdx.x & 31;
  const int half = lane >> 4;
  const int mn   = lane & 15;

  int mrow = tm * 16 + mn; if (mrow >= M) mrow = M - 1;   // clamp for ragged M
  const int ncol0 = tn * 64 + mn;
  const unsigned short* Ar = A + (size_t)mrow * K;
  const unsigned short* Br[4];
#pragma unroll
  for (int t = 0; t < 4; ++t) Br[t] = Bt + (size_t)(ncol0 + 16 * t) * K;
  const int koff = 8 * half;   // 16-bit A/B layout: half-wave selects K groups of 8

  f32x8 acc[4];
#pragma unroll
  for (int t = 0; t < 4; ++t) {
    const float bini = bias ? bias[ncol0 + 16 * t] : 0.f;
#pragma unroll
    for (int r = 0; r < 8; ++r) acc[t][r] = bini;
  }

  for (int k = 0; k < K; k += 32) {
    BFVec a;
    a.u[0] = *(const U128*)(Ar + k + koff);
    a.u[1] = *(const U128*)(Ar + k + 16 + koff);
    __builtin_prefetch(Ar + k + 32 + koff, 0, 1);          // global_prefetch_b8
    BFVec b[4];
#pragma unroll
    for (int t = 0; t < 4; ++t) {
      b[t].u[0] = *(const U128*)(Br[t] + k + koff);
      b[t].u[1] = *(const U128*)(Br[t] + k + 16 + koff);
      __builtin_prefetch(Br[t] + k + 32 + koff, 0, 1);
    }
#pragma unroll
    for (int t = 0; t < 4; ++t)
      acc[t] = __builtin_amdgcn_wmma_f32_16x16x32_bf16(
          false, a.v, false, b[t].v, (short)0, acc[t], false, false);
  }

#pragma unroll
  for (int t = 0; t < 4; ++t) {
    const int nc = ncol0 + 16 * t;
#pragma unroll
    for (int r = 0; r < 8; ++r) {
      const int row = tm * 16 + half * 8 + r;     // C/D layout: VGPR r -> M = 8*half + r
      if (row < M) {
        float x = acc[t][r] * scale;
        if (act == 1) x = gelu_tanh(x);
        const size_t off = (size_t)row * N + nc;
        if (Cf) Cf[off] = x;
        if (Cb) Cb[off] = f2bf(x);
      }
    }
  }
}

// ---------------- weight convert + transpose: dst[c][n][k] = bf16(src[c][k][n]) ----------------
__global__ void conv_transpose(const float* __restrict__ src,
                               unsigned short* __restrict__ dst,
                               int C, int K, int N)
{
  const size_t tot = (size_t)C * K * N;
  size_t idx = (size_t)blockIdx.x * 256 + threadIdx.x;
  if (idx >= tot) return;
  const int k = (int)(idx % K);
  const size_t rem = idx / K;
  const int n = (int)(rem % N);
  const int c = (int)(rem / N);
  dst[idx] = f2bf(src[((size_t)c * K + k) * N + n]);
}

// ---------------- elementwise f32 -> bf16 ----------------
__global__ void to_bf16(const float* __restrict__ in, unsigned short* __restrict__ outp, size_t n)
{
  size_t idx = (size_t)blockIdx.x * 256 + threadIdx.x;
  if (idx < n) outp[idx] = f2bf(in[idx]);
}

// ---------------- embedding sum ----------------
__global__ void embed_kernel(const int* __restrict__ ids, const float* __restrict__ we,
                             const float* __restrict__ pe, float* __restrict__ h)
{
  size_t idx = (size_t)blockIdx.x * 256 + threadIdx.x;
  if (idx >= (size_t)kB * kS * kD) return;
  const int d = (int)(idx % kD);
  const size_t row = idx / kD;
  const int s = (int)(row % kS);
  h[idx] = we[(size_t)ids[row] * kD + d] + pe[(size_t)s * kD + d];
}

// ---------------- LayerNorm(x + resid) * gamma + beta ; one block per row ----------------
__global__ __launch_bounds__(256) void ln_resid(
    const float* __restrict__ x, const float* __restrict__ resid,
    const float* __restrict__ gamma, const float* __restrict__ beta,
    float* __restrict__ outf, unsigned short* __restrict__ outb)
{
  __shared__ float sm[8];
  const size_t row = blockIdx.x;
  const float* xr = x + row * kD;
  const float* rr = resid ? resid + row * kD : nullptr;
  float vals[3]; float s = 0.f;
#pragma unroll
  for (int i = 0; i < 3; ++i) {
    const int d = threadIdx.x + i * 256;
    const float vv = xr[d] + (rr ? rr[d] : 0.f);
    vals[i] = vv; s += vv;
  }
  const float mu = blockSum(s, sm) * (1.f / kD);
  float vsum = 0.f;
#pragma unroll
  for (int i = 0; i < 3; ++i) { const float d0 = vals[i] - mu; vsum += d0 * d0; }
  const float var = blockSum(vsum, sm) * (1.f / kD);
  const float rstd = rsqrtf(var + kEps);
#pragma unroll
  for (int i = 0; i < 3; ++i) {
    const int d = threadIdx.x + i * 256;
    const float o = (vals[i] - mu) * rstd * gamma[d] + beta[d];
    outf[row * kD + d] = o;
    if (outb) outb[row * kD + d] = f2bf(o);
  }
}

// ---------------- find SEP positions / globals ----------------
__global__ void find_sep(const int* __restrict__ ids, const float* __restrict__ mask,
                         int* __restrict__ gpos, float* __restrict__ gkv,
                         unsigned char* __restrict__ isg)
{
  const int b = blockIdx.x;
  if (threadIdx.x != 0) return;
  int gp[kG]; gp[0] = 0; int cnt = 0;
  for (int s = 1; s < kS && cnt < kNSEP; ++s)
    if (ids[b * kS + s] == kSEP) gp[1 + cnt++] = s;
  for (; cnt < kNSEP; ++cnt) gp[1 + cnt] = 0;
  for (int g = 0; g < kG; ++g) {
    gpos[b * kG + g] = gp[g];
    gkv[b * kG + g] = (mask[b * kS + gp[g]] > 0.f) ? 1.f : 0.f;
    isg[b * kS + gp[g]] = 1;
  }
}

// ---------------- gather global-token rows of h into bf16 [B*G, D] ----------------
__global__ void gather_xg(const float* __restrict__ h, const int* __restrict__ gpos,
                          unsigned short* __restrict__ xgb)
{
  const int idx = blockIdx.x * 256 + threadIdx.x;
  if (idx >= kB * kG * kD) return;
  const int d = idx % kD;
  const int row = idx / kD;
  const int b = row / kG;
  const int sp = gpos[row];
  xgb[idx] = f2bf(h[((size_t)(b * kS + sp)) * kD + d]);
}

// ---------------- band (sliding window) + global-key attention ----------------
// one wave per (b, head, query); lane holds dims {2*lane, 2*lane+1} of DH=64
__global__ __launch_bounds__(256) void band_attn(
    const float* __restrict__ q, const float* __restrict__ k, const float* __restrict__ v,
    const float* __restrict__ mask, const unsigned char* __restrict__ isg,
    const int* __restrict__ gpos, const float* __restrict__ gkv,
    float* __restrict__ attn)
{
  const int wid = (int)(((size_t)blockIdx.x * 256 + threadIdx.x) >> 5);
  const int lane = threadIdx.x & 31;
  const int s = wid % kS;
  const int bh = wid / kS;
  const int hh = bh % kH;
  const int b  = bh / kH;

  const size_t qoff = ((size_t)(b * kS + s)) * kD + hh * kDH;
  const float2 qv = *(const float2*)(q + qoff + lane * 2);

  float m = -3.0e38f, l = 0.f;
  float2 acc = make_float2(0.f, 0.f);

  // global keys first (matches concat([global, band]) order; online softmax is order-free)
  for (int g = 0; g < kG; ++g) {
    if (gkv[b * kG + g] <= 0.5f) continue;                  // wave-uniform
    const int j = gpos[b * kG + g];
    const size_t ko = ((size_t)(b * kS + j)) * kD + hh * kDH;
    const float2 kv = *(const float2*)(k + ko + lane * 2);
    const float p = waveSum(qv.x * kv.x + qv.y * kv.y);
    const float mn2 = fmaxf(m, p);
    const float corr = __expf(m - mn2);
    const float w = __expf(p - mn2);
    const float2 vv = *(const float2*)(v + ko + lane * 2);
    l = l * corr + w;
    acc.x = acc.x * corr + w * vv.x;
    acc.y = acc.y * corr + w * vv.y;
    m = mn2;
  }

  // band keys: |j - s| <= W2, valid, not global
  int j0 = s - kW; if (j0 < 0) j0 = 0;
  int j1 = s + kW; if (j1 > kS - 1) j1 = kS - 1;
  for (int j = j0; j <= j1; ++j) {
    if (isg[b * kS + j] || !(mask[b * kS + j] > 0.f)) continue;   // wave-uniform
    const size_t ko = ((size_t)(b * kS + j)) * kD + hh * kDH;
    const float2 kv = *(const float2*)(k + ko + lane * 2);
    const float p = waveSum(qv.x * kv.x + qv.y * kv.y);
    const float mn2 = fmaxf(m, p);
    const float corr = __expf(m - mn2);
    const float w = __expf(p - mn2);
    const float2 vv = *(const float2*)(v + ko + lane * 2);
    l = l * corr + w;
    acc.x = acc.x * corr + w * vv.x;
    acc.y = acc.y * corr + w * vv.y;
    m = mn2;
  }

  const float inv = (l > 0.f) ? 1.f / l : 0.f;
  float* op = attn + qoff + lane * 2;
  op[0] = acc.x * inv;
  op[1] = acc.y * inv;
}

// ---------------- global-token full attention; overwrites attn rows at g_pos ----------------
__global__ __launch_bounds__(256) void glob_attn(
    const float* __restrict__ qg, const float* __restrict__ kg, const float* __restrict__ vg,
    const float* __restrict__ mask, const int* __restrict__ gpos,
    float* __restrict__ attn)
{
  const int wid = (int)(((size_t)blockIdx.x * 256 + threadIdx.x) >> 5);
  if (wid >= kB * kH * kG) return;
  const int lane = threadIdx.x & 31;
  const int g = wid % kG;
  const int bh = wid / kG;
  const int hh = bh % kH;
  const int b  = bh / kH;

  const float2 qv = *(const float2*)(qg + ((size_t)(b * kG + g)) * kD + hh * kDH + lane * 2);
  float m = -3.0e38f, l = 0.f;
  float2 acc = make_float2(0.f, 0.f);

  for (int j = 0; j < kS; ++j) {
    if (!(mask[b * kS + j] > 0.f)) continue;                // wave-uniform
    const size_t ko = ((size_t)(b * kS + j)) * kD + hh * kDH;
    const float2 kv = *(const float2*)(kg + ko + lane * 2);
    const float p = waveSum(qv.x * kv.x + qv.y * kv.y);
    const float mn2 = fmaxf(m, p);
    const float corr = __expf(m - mn2);
    const float w = __expf(p - mn2);
    const float2 vv = *(const float2*)(vg + ko + lane * 2);
    l = l * corr + w;
    acc.x = acc.x * corr + w * vv.x;
    acc.y = acc.y * corr + w * vv.y;
    m = mn2;
  }

  const float inv = (l > 0.f) ? 1.f / l : 0.f;
  const int sp = gpos[b * kG + g];
  float* op = attn + ((size_t)(b * kS + sp)) * kD + hh * kDH + lane * 2;
  op[0] = acc.x * inv;
  op[1] = acc.y * inv;
}

// ---------------- final gather: [cls ; core] per (b, middle-sep) ----------------
__global__ void out_gather(const float* __restrict__ h, const int* __restrict__ gpos,
                           float* __restrict__ outp)
{
  const int r = blockIdx.x;                   // 0 .. B*(NSEP-2)-1
  const int b = r / (kNSEP - 2);
  const int i = r % (kNSEP - 2);
  const int sp = gpos[b * kG + 2 + i];        // core_pos[i] = sep_pos[i+1] = g_pos[i+2]
  for (int d = threadIdx.x; d < kD; d += 256) {
    outp[(size_t)r * 2 * kD + d]      = h[((size_t)b * kS) * kD + d];
    outp[(size_t)r * 2 * kD + kD + d] = h[((size_t)(b * kS + sp)) * kD + d];
  }
}

// =======================================================================
extern "C" void kernel_launch(void* const* d_in, const int* in_sizes, int n_in,
                              void* d_out, int out_size, void* d_ws, size_t ws_size,
                              hipStream_t stream)
{
  (void)in_sizes; (void)n_in; (void)out_size; (void)ws_size;
  const int*   ids   = (const int*)d_in[0];
  const float* mask  = (const float*)d_in[1];
  const float* wemb  = (const float*)d_in[2];
  const float* pemb  = (const float*)d_in[3];
  const float* embln = (const float*)d_in[4];   // [2, D]
  const float* Wqkv  = (const float*)d_in[5];   // [L,3,D,D]
  const float* bqkv  = (const float*)d_in[6];   // [L,3,D]
  const float* Wg    = (const float*)d_in[7];   // [L,3,D,D]
  const float* bg    = (const float*)d_in[8];   // [L,3,D]
  const float* Wo    = (const float*)d_in[9];   // [L,D,D]
  const float* bo    = (const float*)d_in[10];  // [L,D]
  const float* ln1   = (const float*)d_in[11];  // [2,L,D]
  const float* W1    = (const float*)d_in[12];  // [L,D,FF]
  const float* b1    = (const float*)d_in[13];  // [L,FF]
  const float* W2m   = (const float*)d_in[14];  // [L,FF,D]
  const float* b2    = (const float*)d_in[15];  // [L,D]
  const float* ln2   = (const float*)d_in[16];  // [2,L,D]
  float* outp = (float*)d_out;

  // ---- workspace carve-up ----
  char* p = (char*)d_ws;
  auto alloc = [&](size_t bytes) -> char* {
    char* r = p; p += (bytes + 255) & ~(size_t)255; return r;
  };
  const size_t BS  = (size_t)kB * kS;
  const size_t BSD = BS * kD;

  int*            gpos  = (int*)alloc((size_t)kB * kG * 4);
  float*          gkv   = (float*)alloc((size_t)kB * kG * 4);
  unsigned char*  isg   = (unsigned char*)alloc(BS);
  unsigned short* WqkvT = (unsigned short*)alloc((size_t)kL * 3 * kD * kD * 2);
  unsigned short* WgT   = (unsigned short*)alloc((size_t)kL * 3 * kD * kD * 2);
  unsigned short* WoT   = (unsigned short*)alloc((size_t)kL * kD * kD * 2);
  unsigned short* W1T   = (unsigned short*)alloc((size_t)kL * kD * kFF * 2);   // [l][FF][D]
  unsigned short* W2T   = (unsigned short*)alloc((size_t)kL * kFF * kD * 2);   // [l][D][FF]
  float*          h     = (float*)alloc(BSD * 4);
  unsigned short* hb    = (unsigned short*)alloc(BSD * 2);
  float*          q     = (float*)alloc(BSD * 4);
  float*          kk    = (float*)alloc(BSD * 4);
  float*          vv    = (float*)alloc(BSD * 4);
  float*          kg    = (float*)alloc(BSD * 4);   // kg+vg region == t1 bf16 region (sizes match)
  float*          vg    = (float*)alloc(BSD * 4);
  float*          qg    = (float*)alloc((size_t)kB * kG * kD * 4);
  unsigned short* xgb   = (unsigned short*)alloc((size_t)kB * kG * kD * 2);
  float*          attn  = (float*)alloc(BSD * 4);
  unsigned short* attnb = (unsigned short*)alloc(BSD * 2);
  // reuse: o := q, t2 := kk, t1b := kg region (B*S*FF*2 == 2*B*S*D*4 bytes)
  float*          o_buf = q;
  float*          t2    = kk;
  unsigned short* t1b   = (unsigned short*)kg;

  auto gemm = [&](const unsigned short* A, const unsigned short* Bt, const float* bias,
                  float* Cf, unsigned short* Cb, int M, int N, int K, float scale, int act) {
    const int tiles = ((M + 15) / 16) * (N / 64);   // one wave -> 16x64 tile
    gemm_bf16<<<(tiles + 7) / 8, 256, 0, stream>>>(A, Bt, bias, Cf, Cb, M, N, K, scale, act);
  };
  auto cvt = [&](const float* src, unsigned short* dst, size_t n) {
    to_bf16<<<(int)((n + 255) / 256), 256, 0, stream>>>(src, dst, n);
  };

  // ---- one-time per call: masks / globals / weight prep ----
  hipMemsetAsync(isg, 0, BS, stream);
  find_sep<<<kB, 32, 0, stream>>>(ids, mask, gpos, gkv, isg);

  {
    const size_t nqkv = (size_t)kL * 3 * kD * kD;
    conv_transpose<<<(int)((nqkv + 255) / 256), 256, 0, stream>>>(Wqkv, WqkvT, kL * 3, kD, kD);
    conv_transpose<<<(int)((nqkv + 255) / 256), 256, 0, stream>>>(Wg,   WgT,   kL * 3, kD, kD);
    const size_t nwo = (size_t)kL * kD * kD;
    conv_transpose<<<(int)((nwo + 255) / 256), 256, 0, stream>>>(Wo, WoT, kL, kD, kD);
    const size_t nff = (size_t)kL * kD * kFF;
    conv_transpose<<<(int)((nff + 255) / 256), 256, 0, stream>>>(W1,  W1T, kL, kD, kFF);
    conv_transpose<<<(int)((nff + 255) / 256), 256, 0, stream>>>(W2m, W2T, kL, kFF, kD);
  }

  // ---- embeddings + LN ----
  embed_kernel<<<(int)((BSD + 255) / 256), 256, 0, stream>>>(ids, wemb, pemb, h);
  ln_resid<<<(int)BS, 256, 0, stream>>>(h, nullptr, embln, embln + kD, h, hb);

  const int M = (int)BS;   // 8192 token rows
  const int DD = kD * kD;

  for (int l = 0; l < kL; ++l) {
    // QKV projections (bf16 WMMA, fp32 out)
    gemm(hb, WqkvT + (size_t)(l * 3 + 0) * DD, bqkv + (size_t)(l * 3 + 0) * kD,
         q,  nullptr, M, kD, kD, kScale, 0);
    gemm(hb, WqkvT + (size_t)(l * 3 + 1) * DD, bqkv + (size_t)(l * 3 + 1) * kD,
         kk, nullptr, M, kD, kD, 1.f, 0);
    gemm(hb, WqkvT + (size_t)(l * 3 + 2) * DD, bqkv + (size_t)(l * 3 + 2) * kD,
         vv, nullptr, M, kD, kD, 1.f, 0);
    // global-attention K/V over full sequence (Wg[l,1], Wg[l,2])
    gemm(hb, WgT + (size_t)(l * 3 + 1) * DD, bg + (size_t)(l * 3 + 1) * kD,
         kg, nullptr, M, kD, kD, 1.f, 0);
    gemm(hb, WgT + (size_t)(l * 3 + 2) * DD, bg + (size_t)(l * 3 + 2) * kD,
         vg, nullptr, M, kD, kD, 1.f, 0);
    // global queries: gather h at g_pos -> bf16, tiny GEMM (M = B*G = 18)
    gather_xg<<<(kB * kG * kD + 255) / 256, 256, 0, stream>>>(h, gpos, xgb);
    gemm(xgb, WgT + (size_t)(l * 3 + 0) * DD, bg + (size_t)(l * 3 + 0) * kD,
         qg, nullptr, kB * kG, kD, kD, kScale, 0);

    // band + global-key attention (one wave per query row)
    band_attn<<<(kB * kH * kS) / 8, 256, 0, stream>>>(q, kk, vv, mask, isg, gpos, gkv, attn);
    // full attention for global tokens; overwrites attn rows at g_pos
    glob_attn<<<(kB * kH * kG * 32 + 255) / 256, 256, 0, stream>>>(qg, kg, vg, mask, gpos, attn);

    // output projection + residual + LN1
    cvt(attn, attnb, BSD);
    gemm(attnb, WoT + (size_t)l * DD, bo + (size_t)l * kD, o_buf, nullptr, M, kD, kD, 1.f, 0);
    ln_resid<<<(int)BS, 256, 0, stream>>>(h, o_buf,
        ln1 + ((size_t)l * kL + 0) * kD, ln1 + ((size_t)l * kL + 1) * kD, h, hb);

    // FFN: gelu(h@W1+b1)@W2+b2, residual + LN2
    gemm(hb,  W1T + (size_t)l * kD * kFF, b1 + (size_t)l * kFF,
         nullptr, t1b, M, kFF, kD, 1.f, 1);
    gemm(t1b, W2T + (size_t)l * kFF * kD, b2 + (size_t)l * kD,
         t2, nullptr, M, kD, kFF, 1.f, 0);
    ln_resid<<<(int)BS, 256, 0, stream>>>(h, t2,
        ln2 + ((size_t)l * kL + 0) * kD, ln2 + ((size_t)l * kL + 1) * kD, h, hb);
  }

  out_gather<<<kB * (kNSEP - 2), 256, 0, stream>>>(h, gpos, outp);
}
